// OrcaOPTDecoderLayer_46462956208351
// MI455X (gfx1250) — matrix operand order repaired and checked by reading, not compiled, not run on hardware
//
#include <hip/hip_runtime.h>
#include <hip/hip_bf16.h>

typedef __attribute__((ext_vector_type(16))) _Float16 v16h;
typedef __attribute__((ext_vector_type(8)))  _Float16 v8h;
typedef __attribute__((ext_vector_type(8)))  float    v8f;
typedef __attribute__((ext_vector_type(2)))  _Float16 hpair;
typedef __attribute__((ext_vector_type(4)))  int      v4i;

#define TOTAL 6144
#define EMBED 768
#define NHEAD 12
#define HDIM  64
#define FFN   3072

__device__ __constant__ int SOFF[9] = {0,1024,1792,2304,3200,3840,4224,5248,6144};
__device__ __constant__ int QBC[9]  = {0,8,14,18,25,30,33,41,48};   // cum qblocks of 128

static __device__ inline v8f wmma16(v16h a, v16h b, v8f c) {
    return __builtin_amdgcn_wmma_f32_16x16x32_f16(false, a, false, b, (short)0, c, false, false);
}
static __device__ inline float rmax16(float x) {
    #pragma unroll
    for (int o = 1; o < 16; o <<= 1) x = fmaxf(x, __shfl_xor(x, o, 32));
    return x;
}
static __device__ inline float rsum16(float x) {
    #pragma unroll
    for (int o = 1; o < 16; o <<= 1) x += __shfl_xor(x, o, 32);
    return x;
}

// ---------------- async global->LDS (CDNA5 GLOBAL_LOAD_ASYNC_TO_LDS_B128) ---
#if __has_builtin(__builtin_amdgcn_global_load_async_to_lds_b128) && \
    __has_builtin(__builtin_amdgcn_s_wait_asynccnt)
#define HAVE_ASYNC_LDS 1
typedef __attribute__((address_space(1))) v4i g_v4i;
typedef __attribute__((address_space(3))) v4i l_v4i;
static __device__ inline void cp16(uint4* dst_lds, const uint4* src_g) {
    __builtin_amdgcn_global_load_async_to_lds_b128(
        (g_v4i*)(void*)src_g, (l_v4i*)(void*)dst_lds, 0, 0);
}
#define TILE_WAIT() __builtin_amdgcn_s_wait_asynccnt(0)
#else
#define HAVE_ASYNC_LDS 0
static __device__ inline void cp16(uint4* dst_lds, const uint4* src_g) {
    *dst_lds = *src_g;   // global_load_b128 + ds_store_b128
}
#define TILE_WAIT() ((void)0)
#endif

// ---------------------------------------------------------------- fp32->fp16
__global__ __launch_bounds__(256) void cvt_kernel(const float* __restrict__ s,
                                                  _Float16* __restrict__ d, int n) {
    for (int i = blockIdx.x * 256 + threadIdx.x; i < n; i += gridDim.x * 256)
        d[i] = (_Float16)s[i];
}

// ---------------------------------------------------------------- LayerNorm
__global__ __launch_bounds__(256) void ln_kernel(const float* __restrict__ x,
                                                 const float* __restrict__ g,
                                                 const float* __restrict__ b,
                                                 _Float16* __restrict__ out) {
    __shared__ float red[256];
    const int row = blockIdx.x, t = threadIdx.x;
    const float* xr = x + (size_t)row * EMBED;
    float v0 = xr[t], v1 = xr[t + 256], v2 = xr[t + 512];
    red[t] = v0 + v1 + v2;
    __syncthreads();
    #pragma unroll
    for (int o = 128; o > 0; o >>= 1) { if (t < o) red[t] += red[t + o]; __syncthreads(); }
    const float mu = red[0] * (1.0f / EMBED);
    __syncthreads();
    const float d0 = v0 - mu, d1 = v1 - mu, d2 = v2 - mu;
    red[t] = d0 * d0 + d1 * d1 + d2 * d2;
    __syncthreads();
    #pragma unroll
    for (int o = 128; o > 0; o >>= 1) { if (t < o) red[t] += red[t + o]; __syncthreads(); }
    const float rs = rsqrtf(red[0] * (1.0f / EMBED) + 1e-5f);
    _Float16* orow = out + (size_t)row * EMBED;
    orow[t]       = (_Float16)(d0 * rs * g[t]       + b[t]);
    orow[t + 256] = (_Float16)(d1 * rs * g[t + 256] + b[t + 256]);
    orow[t + 512] = (_Float16)(d2 * rs * g[t + 512] + b[t + 512]);
}

// --------------------------------------------- generic WMMA GEMM: C=A*W^T+eps
// A[M,K] f16 row-major, W[N,K] f16 row-major. Block: 256 thr = 8 waves;
// tile 32(M) x 256(N); K-step 32; double-buffered LDS, async tile streaming.
template<bool RELU, bool OUT16>
__global__ __launch_bounds__(256)
void gemm_kernel(const _Float16* __restrict__ A, const _Float16* __restrict__ W,
                 const float* __restrict__ bias, const float* __restrict__ resid,
                 void* __restrict__ out, int M, int N, int K) {
    __shared__ uint4 aT[2][32 * 4];     // 32 rows x 32 halfs
    __shared__ uint4 bT[2][256 * 4];    // 256 rows x 32 halfs
    const unsigned tid = threadIdx.x;
    const unsigned lane = tid & 31u, lo = lane & 15u, hi = lane >> 4;
    const unsigned w = tid >> 5, wm = w >> 2, wn = w & 3u;
    const unsigned nBase = blockIdx.x * 256u, mBase = blockIdx.y * 32u;
    const unsigned Ku4 = (unsigned)K >> 3;              // uint4 per row
    const uint4* __restrict__ Ag4 = (const uint4*)A;
    const uint4* __restrict__ Wg4 = (const uint4*)W;

    auto issue_tile = [&](int k0, int b) {
        const unsigned kq = (unsigned)k0 >> 3;
        if (tid < 128u) {                               // A tile: 128 uint4
            const unsigned r = tid >> 2, c = tid & 3u;
            cp16(&aT[b][tid], Ag4 + (size_t)(mBase + r) * Ku4 + kq + c);
        }
        #pragma unroll
        for (int i = 0; i < 4; ++i) {                   // B tile: 1024 uint4
            const unsigned idx = tid + i * 256u, r = idx >> 2, c = idx & 3u;
            cp16(&bT[b][idx], Wg4 + (size_t)(nBase + r) * Ku4 + kq + c);
        }
    };

    v8f c0{}, c1{}, c2{}, c3{};
    issue_tile(0, 0);
    const int nk = K >> 5;
    for (int t = 0; t < nk; ++t) {
        TILE_WAIT();
        __syncthreads();
        if (t + 1 < nk) issue_tile((t + 1) << 5, (t + 1) & 1);
        const unsigned* aTu = (const unsigned*)aT[t & 1];
        const unsigned* bTu = (const unsigned*)bT[t & 1];
        v16h af;                        // A frag: lane row = wm*16+lo
        #pragma unroll
        for (int v = 0; v < 8; ++v) {
            const unsigned ku = hi * 8u + (unsigned)v;
            hpair p = __builtin_bit_cast(hpair, aTu[(wm * 16u + lo) * 16u + ku]);
            af[2 * v] = p.x; af[2 * v + 1] = p.y;
        }
        #pragma unroll
        for (int s = 0; s < 4; ++s) {   // B frag: lane col = wn*64+s*16+lo
            v16h bf;
            const unsigned nl = wn * 64u + s * 16u + lo;
            #pragma unroll
            for (int v = 0; v < 8; ++v) {
                hpair p = __builtin_bit_cast(hpair, bTu[nl * 16u + hi * 8u + (unsigned)v]);
                bf[2 * v] = p.x; bf[2 * v + 1] = p.y;
            }
            if (s == 0) c0 = wmma16(af, bf, c0);
            else if (s == 1) c1 = wmma16(af, bf, c1);
            else if (s == 2) c2 = wmma16(af, bf, c2);
            else             c3 = wmma16(af, bf, c3);
        }
    }
    #pragma unroll
    for (int s = 0; s < 4; ++s) {
        v8f cc = (s == 0) ? c0 : (s == 1) ? c1 : (s == 2) ? c2 : c3;
        const unsigned col = nBase + wn * 64u + s * 16u + lo;
        const float bv = bias ? bias[col] : 0.0f;
        #pragma unroll
        for (int v = 0; v < 8; ++v) {
            const unsigned row = mBase + wm * 16u + (unsigned)v + 8u * hi;
            float val = cc[v] + bv;
            if (RELU) val = fmaxf(val, 0.0f);
            if (resid) val += resid[(size_t)row * N + col];
            if (OUT16) ((_Float16*)out)[(size_t)row * N + col] = (_Float16)val;
            else       ((float*)out)[(size_t)row * N + col] = val;
        }
    }
}

// ------------------------------------- flash-style causal attention (ragged)
// qkv: [TOTAL, 3*EMBED] f16 (Q | K | V). One block = 128 queries of one head.
__global__ __launch_bounds__(256)
void attn_kernel(const _Float16* __restrict__ qkv, _Float16* __restrict__ ctx) {
    __shared__ uint4     kT4[32 * 8];       // [key][64 halfs] K tile (d-major)
    __shared__ _Float16  vT[64 * 32];       // [d][key]        V tile transposed
    __shared__ _Float16  pT[8 * 16 * 32];   // per-wave P re-layout buffer
    const int bid = blockIdx.x;
    const int h = bid % NHEAD;
    const int qidx = bid / NHEAD;
    int b = 0;
    #pragma unroll
    for (int i = 0; i < 8; ++i) if (qidx >= QBC[i + 1]) b = i + 1;
    const int soff = SOFF[b];
    const int qstart = (qidx - QBC[b]) * 128;
    const int tid = threadIdx.x;
    const int lane = tid & 31, lo = lane & 15, hi = lane >> 4;
    const int wv = tid >> 5;
    const uint4* __restrict__ qg4 = (const uint4*)qkv;  // 288 uint4/token

    // Q fragments (d 0..31 and 32..63), lane row = lo : 4x b128 loads
    const size_t qb4 = (size_t)(soff + qstart + wv * 16 + lo) * 288 + (size_t)h * 8;
    v16h qf0, qf1;
    {
        const uint4 u0 = qg4[qb4 + 2 * hi];     // u32 [hi*8 .. hi*8+3]
        const uint4 u1 = qg4[qb4 + 2 * hi + 1]; // u32 [hi*8+4 ..]
        const uint4 u2 = qg4[qb4 + 4 + 2 * hi]; // d+32 halves
        const uint4 u3 = qg4[qb4 + 5 + 2 * hi];
        const v8h h0 = __builtin_bit_cast(v8h, u0), h1 = __builtin_bit_cast(v8h, u1);
        const v8h h2 = __builtin_bit_cast(v8h, u2), h3 = __builtin_bit_cast(v8h, u3);
        #pragma unroll
        for (int j = 0; j < 8; ++j) {
            qf0[j] = h0[j]; qf0[8 + j] = h1[j];
            qf1[j] = h2[j]; qf1[8 + j] = h3[j];
        }
    }
    v8f o0{}, o1{}, o2{}, o3{};
    float mr[8], lr[8];
    #pragma unroll
    for (int v = 0; v < 8; ++v) { mr[v] = -3.0e38f; lr[v] = 0.0f; }

    const int nt = (qstart + 128) >> 5;   // causal: keys < qend, 32-key tiles
    for (int t = 0; t < nt; ++t) {
        // prefetch this tile's K/V into registers (overlaps prev compute)
        const int key = tid >> 3, c = tid & 7;      // one uint4 each
        const size_t tok4 = (size_t)(soff + t * 32 + key) * 288 + (size_t)h * 8;
        const uint4 kreg = qg4[tok4 + 96 + c];      // K: +768 halfs
        const uint4 vreg = qg4[tok4 + 192 + c];     // V: +1536 halfs
        __syncthreads();
        kT4[key * 8 + c] = kreg;
        const v8h vh = __builtin_bit_cast(v8h, vreg);
        #pragma unroll
        for (int j = 0; j < 8; ++j) vT[(c * 8 + j) * 32 + key] = vh[j];
        __syncthreads();

        // S = Q * K^T  (two 16-col subtiles; contraction d split 0-31/32-63)
        const unsigned* kTu = (const unsigned*)kT4;
        v8f s0{}, s1{};
        {
            v16h ba, bb;
            #pragma unroll
            for (int v = 0; v < 8; ++v) {
                hpair pa = __builtin_bit_cast(hpair, kTu[lo * 32 + hi * 8 + v]);
                hpair pb = __builtin_bit_cast(hpair, kTu[lo * 32 + 16 + hi * 8 + v]);
                ba[2 * v] = pa.x; ba[2 * v + 1] = pa.y;
                bb[2 * v] = pb.x; bb[2 * v + 1] = pb.y;
            }
            s0 = wmma16(qf0, ba, s0); s0 = wmma16(qf1, bb, s0);
            #pragma unroll
            for (int v = 0; v < 8; ++v) {
                hpair pa = __builtin_bit_cast(hpair, kTu[(16 + lo) * 32 + hi * 8 + v]);
                hpair pb = __builtin_bit_cast(hpair, kTu[(16 + lo) * 32 + 16 + hi * 8 + v]);
                ba[2 * v] = pa.x; ba[2 * v + 1] = pa.y;
                bb[2 * v] = pb.x; bb[2 * v + 1] = pb.y;
            }
            s1 = wmma16(qf0, ba, s1); s1 = wmma16(qf1, bb, s1);
        }
        // scale + causal mask + online softmax (per-VGPR row m = v+8*hi)
        const int kb = t * 32;
        #pragma unroll
        for (int v = 0; v < 8; ++v) {
            const int qi = qstart + wv * 16 + v + 8 * hi;
            s0[v] = (kb + lo      <= qi) ? s0[v] * 0.125f : -1.0e30f;
            s1[v] = (kb + 16 + lo <= qi) ? s1[v] * 0.125f : -1.0e30f;
        }
        #pragma unroll
        for (int v = 0; v < 8; ++v) {
            const float mx = rmax16(fmaxf(s0[v], s1[v]));
            const float mn = fmaxf(mr[v], mx);
            const float alpha = __expf(mr[v] - mn);
            const float p0 = __expf(s0[v] - mn);
            const float p1 = __expf(s1[v] - mn);
            lr[v] = lr[v] * alpha + rsum16(p0 + p1);
            mr[v] = mn;
            o0[v] *= alpha; o1[v] *= alpha; o2[v] *= alpha; o3[v] *= alpha;
            const int m = v + 8 * hi;
            pT[wv * 512 + m * 32 + lo]      = (_Float16)p0;
            pT[wv * 512 + m * 32 + 16 + lo] = (_Float16)p1;
        }
        // re-layout P (C-layout -> A-layout) through per-wave LDS region
        v16h pa;
        const unsigned* pTu = (const unsigned*)pT;
        #pragma unroll
        for (int v = 0; v < 8; ++v) {
            const int ku = hi * 8 + v;
            hpair p = __builtin_bit_cast(hpair, pTu[wv * 256 + lo * 16 + ku]);
            pa[2 * v] = p.x; pa[2 * v + 1] = p.y;
        }
        // O += P * V (contraction = 32 keys; 4 d-subtiles of 16)
        const unsigned* vTu = (const unsigned*)vT;
        #pragma unroll
        for (int j = 0; j < 4; ++j) {
            v16h bv;
            const int d = j * 16 + lo;
            #pragma unroll
            for (int v = 0; v < 8; ++v) {
                hpair p = __builtin_bit_cast(hpair, vTu[d * 16 + hi * 8 + v]);
                bv[2 * v] = p.x; bv[2 * v + 1] = p.y;
            }
            if (j == 0) o0 = wmma16(pa, bv, o0);
            else if (j == 1) o1 = wmma16(pa, bv, o1);
            else if (j == 2) o2 = wmma16(pa, bv, o2);
            else             o3 = wmma16(pa, bv, o3);
        }
    }
    // normalize + write ctx [TOTAL, EMBED] f16
    #pragma unroll
    for (int j = 0; j < 4; ++j) {
        v8f oo = (j == 0) ? o0 : (j == 1) ? o1 : (j == 2) ? o2 : o3;
        #pragma unroll
        for (int v = 0; v < 8; ++v) {
            const int token = soff + qstart + wv * 16 + v + 8 * hi;
            const int col = h * 64 + j * 16 + lo;
            ctx[(size_t)token * EMBED + col] = (_Float16)(oo[v] / lr[v]);
        }
    }
}

// ---------------------------------------------------------------- launcher
extern "C" void kernel_launch(void* const* d_in, const int* in_sizes, int n_in,
                              void* d_out, int out_size, void* d_ws, size_t ws_size,
                              hipStream_t stream) {
    (void)in_sizes; (void)n_in; (void)out_size; (void)ws_size;
    const float* hidden    = (const float*)d_in[0];
    const float* in_proj_w = (const float*)d_in[2];
    const float* in_proj_b = (const float*)d_in[3];
    const float* out_proj_w= (const float*)d_in[4];
    const float* out_proj_b= (const float*)d_in[5];
    const float* ln1_g     = (const float*)d_in[6];
    const float* ln1_b     = (const float*)d_in[7];
    const float* fc1_w     = (const float*)d_in[8];
    const float* fc1_b     = (const float*)d_in[9];
    const float* fc2_w     = (const float*)d_in[10];
    const float* fc2_b     = (const float*)d_in[11];
    const float* ln2_g     = (const float*)d_in[12];
    const float* ln2_b     = (const float*)d_in[13];

    char* ws = (char*)d_ws;
    _Float16* xln  = (_Float16*)(ws + 0);                    //  9,437,184 (reused as h_ln)
    _Float16* wqkv = (_Float16*)(ws + 9437184);              //  3,538,944
    _Float16* wout = (_Float16*)(ws + 12976128);             //  1,179,648
    _Float16* w1   = (_Float16*)(ws + 14155776);             //  4,718,592
    _Float16* w2   = (_Float16*)(ws + 18874368);             //  4,718,592
    _Float16* qkvb = (_Float16*)(ws + 23592960);             // 28,311,552
    _Float16* ctxb = (_Float16*)(ws + 51904512);             //  9,437,184
    float*    hbuf = (float*)   (ws + 61341696);             // 18,874,368
    _Float16* ffn1 = (_Float16*)(ws + 23592960);             // aliases qkvb+ctxb (consumed)

    cvt_kernel<<<2048, 256, 0, stream>>>(in_proj_w, wqkv, 3 * EMBED * EMBED);
    cvt_kernel<<<2048, 256, 0, stream>>>(out_proj_w, wout, EMBED * EMBED);
    cvt_kernel<<<2048, 256, 0, stream>>>(fc1_w, w1, FFN * EMBED);
    cvt_kernel<<<2048, 256, 0, stream>>>(fc2_w, w2, EMBED * FFN);
    ln_kernel<<<TOTAL, 256, 0, stream>>>(hidden, ln1_g, ln1_b, xln);
    gemm_kernel<false, true><<<dim3(9, 192), 256, 0, stream>>>(
        xln, wqkv, in_proj_b, nullptr, qkvb, TOTAL, 3 * EMBED, EMBED);
    attn_kernel<<<48 * NHEAD, 256, 0, stream>>>(qkvb, ctxb);
    gemm_kernel<false, false><<<dim3(3, 192), 256, 0, stream>>>(
        ctxb, wout, out_proj_b, hidden, hbuf, TOTAL, EMBED, EMBED);
    ln_kernel<<<TOTAL, 256, 0, stream>>>(hbuf, ln2_g, ln2_b, xln);
    gemm_kernel<true, true><<<dim3(12, 192), 256, 0, stream>>>(
        xln, w1, fc1_b, nullptr, ffn1, TOTAL, FFN, EMBED);
    gemm_kernel<false, false><<<dim3(3, 192), 256, 0, stream>>>(
        ffn1, w2, fc2_b, hbuf, (float*)d_out, TOTAL, EMBED, FFN);
}
